// Retention_19284403159613
// MI455X (gfx1250) — matrix operand order repaired and checked by reading, not compiled, not run on hardware
//
#include <hip/hip_runtime.h>

// RetNet retention, B=4, S=4096, D=256, gamma=0.96875.
// Kernel 1: projections via bf16 WMMA -> Q,K row-major bf16; V transposed.
// Kernel 2: retention; K/V blocks staged to LDS by the Tensor Data Mover
//           (tensor_load_to_lds, double-buffered, TENSORcnt-synchronized),
//           scores*decay reswizzled C->A layout through LDS, all matmuls
//           v_wmma_f32_16x16x32_bf16 with f32 accumulation.

#define S_LEN  4096
#define BATCH  4
#define DMODEL 256
#define BS     (BATCH * S_LEN)

// log2(0.96875)
#define L2G (-0.0458039140f)

typedef __attribute__((ext_vector_type(16))) __bf16 bf16x16;
typedef __attribute__((ext_vector_type(8)))  __bf16 bf16x8;
typedef __attribute__((ext_vector_type(8)))  float  f32x8;
typedef unsigned int u32x4 __attribute__((ext_vector_type(4)));
typedef int          i32x8 __attribute__((ext_vector_type(8)));
typedef int          i32x4 __attribute__((ext_vector_type(4)));

__device__ __forceinline__ bf16x16 ld_frag(const __bf16* p, int gap) {
  bf16x8 a = *(const bf16x8*)p;
  bf16x8 b = *(const bf16x8*)(p + gap);
  return __builtin_shufflevector(a, b, 0,1,2,3,4,5,6,7,8,9,10,11,12,13,14,15);
}

__device__ __forceinline__ f32x8 wmma_bf16(bf16x16 a, bf16x16 b, f32x8 c) {
  return __builtin_amdgcn_wmma_f32_16x16x32_bf16(false, a, false, b,
                                                 (short)0, c, false, false);
}

// --- Tensor Data Mover: 2D tile load, 2-byte elements (ISA 08, D# groups) ---
// Generic pointers to LDS carry the LDS byte offset in their low 32 bits
// (aperture in the high bits), so (u32)ptr is the D#.lds_addr.
// This toolchain's builtin takes 6 args: g0(x4), g1(x8), g2(x4), g3(x4),
// extra(x8), cpol. Groups 2/3 + extra are unused for a 2D tile -> zeros.
__device__ __forceinline__ void tdm_load_2d(const void* lds_dst, uint64_t gaddr,
                                            uint32_t tdim0, uint32_t tdim1,
                                            uint32_t tile0, uint32_t tile1,
                                            uint64_t stride0) {
  u32x4 g0;
  g0.x = 1u;                                                  // count=1, user D#
  g0.y = (uint32_t)(uintptr_t)lds_dst;                        // lds_addr[31:0]
  g0.z = (uint32_t)gaddr;                                     // global_addr[31:0]
  g0.w = ((uint32_t)(gaddr >> 32) & 0x1FFFFFFu) | (2u << 30); // addr[56:32], type=2
  i32x8 g1 = (i32x8)0;
  g1.s0 = (int)(1u << 16);                                    // data_size = 2B
  g1.s1 = (int)((tdim0 & 0xFFFFu) << 16);                     // tensor_dim0[15:0]
  g1.s2 = (int)(((tdim0 >> 16) & 0xFFFFu) | ((tdim1 & 0xFFFFu) << 16));
  g1.s3 = (int)(((tdim1 >> 16) & 0xFFFFu) | ((tile0 & 0xFFFFu) << 16));
  g1.s4 = (int)(tile1 & 0xFFFFu);                             // tile_dim1 (dim2=0)
  g1.s5 = (int)(uint32_t)stride0;                             // dim0_stride[31:0]
  g1.s6 = (int)((uint32_t)(stride0 >> 32) & 0xFFFFu);         // dim0_stride[47:32]
  g1.s7 = 0;
  i32x4 z4 = (i32x4)0;
  i32x8 z8 = (i32x8)0;
  __builtin_amdgcn_tensor_load_to_lds(g0, g1, z4, z4, z8, 0);
}

__device__ __forceinline__ void wait_tensorcnt0() {
#if __has_builtin(__builtin_amdgcn_s_wait_tensorcnt)
  __builtin_amdgcn_s_wait_tensorcnt(0);
#else
  asm volatile("s_wait_tensorcnt 0x0" ::: "memory");
#endif
}

// ---------------------------------------------------------------------------
// Kernel 1: projections. One wave per 16x16 output tile per matrix.
// y = x @ W^T :  A[m][k] = x[m][k],  B[k][n] = W[n][k]
// ---------------------------------------------------------------------------
__global__ void __launch_bounds__(128)
retnet_proj(const float* __restrict__ x,
            const float* __restrict__ Wq,
            const float* __restrict__ Wk,
            const float* __restrict__ Wv,
            __bf16* __restrict__ Qb,   // [BS][D] row-major
            __bf16* __restrict__ Kb,   // [BS][D] row-major
            __bf16* __restrict__ Vt)   // [D][BS] (transposed)
{
  const int lane = threadIdx.x & 31;
  const int wave = threadIdx.x >> 5;
  const int lo   = lane & 15;
  const int hi   = lane >> 4;

  int gw  = blockIdx.x * 4 + wave;
  int nt  = gw & 15;
  int t   = gw >> 4;
  int mt  = t & 1023;
  int mat = t >> 10;                        // 0=Q, 1=K, 2=V

  const float* W = (mat == 0) ? Wq : (mat == 1) ? Wk : Wv;
  const int m0 = mt * 16, n0 = nt * 16;

  f32x8 acc = {};
  const float* arow = x + (size_t)(m0 + lo) * DMODEL;
  const float* brow = W + (size_t)(n0 + lo) * DMODEL;
#pragma unroll
  for (int k0 = 0; k0 < DMODEL; k0 += 32) {
    bf16x16 a, b;
    const float* ap = arow + k0 + hi * 16;
    const float* bp = brow + k0 + hi * 16;
#pragma unroll
    for (int e = 0; e < 8; ++e) { a[e] = (__bf16)ap[e]; a[e + 8] = (__bf16)ap[e + 8]; }
#pragma unroll
    for (int e = 0; e < 16; ++e) b[e] = (__bf16)bp[e];
    acc = wmma_bf16(a, b, acc);
  }

  if (mat < 2) {
    __bf16* out = (mat == 0) ? Qb : Kb;
#pragma unroll
    for (int r = 0; r < 8; ++r)
      out[(size_t)(m0 + r + hi * 8) * DMODEL + n0 + lo] = (__bf16)acc[r];
  } else {
    __bf16* vp = Vt + (size_t)(n0 + lo) * BS + m0 + hi * 8;
    bf16x8 pk;
#pragma unroll
    for (int r = 0; r < 8; ++r) pk[r] = (__bf16)acc[r];
    *(bf16x8*)vp = pk;
  }
}

// ---------------------------------------------------------------------------
// Kernel 2: retention. Block = 4 waves = 64 query rows; all waves iterate the
// same (uniform) key-block count; decay mask zeroes non-causal blocks.
// K/V blocks double-buffered in LDS via TDM, issued by wave 0.
// ---------------------------------------------------------------------------
__global__ void __launch_bounds__(128)
retnet_attn(const __bf16* __restrict__ Qb,
            const __bf16* __restrict__ Kb,
            const __bf16* __restrict__ Vt,
            float* __restrict__ out)
{
  const int lane = threadIdx.x & 31;
  const int wave = threadIdx.x >> 5;
  const int lo   = lane & 15;
  const int hi   = lane >> 4;

  const int wg   = blockIdx.x;               // BATCH * S/64 = 256
  const int b    = wg >> 6;
  const int q64  = wg & 63;
  const int i0   = q64 * 64 + wave * 16;     // this wave's query tile
  const int nblk = 2 * q64 + 2;              // uniform key-block count

  __shared__ __align__(16) __bf16 ldsK[2][32 * DMODEL];  // K block, row-major
  __shared__ __align__(16) __bf16 ldsV[2][DMODEL * 32];  // V^T block: [d][k]
  __shared__ __align__(16) __bf16 sc[4][16 * 32];        // per-wave scores

  // Resident Q fragments (8 chunks of K=32)
  bf16x16 qf[8];
  const __bf16* qrow = Qb + (size_t)(b * S_LEN + i0 + lo) * DMODEL + hi * 16;
#pragma unroll
  for (int c = 0; c < 8; ++c) qf[c] = ld_frag(qrow + c * 32, 8);

  // Per-lane decay factors gamma^(M-N), M = r+8*hi, N = lo
  float ld8[8];
#pragma unroll
  for (int r = 0; r < 8; ++r)
    ld8[r] = exp2f((float)((r + hi * 8) - lo) * L2G);

  f32x8 acc[16];
#pragma unroll
  for (int n = 0; n < 16; ++n) acc[n] = (f32x8){};

  const uint64_t kaddr = (uint64_t)(uintptr_t)(Kb + (size_t)(b * S_LEN) * DMODEL);
  const uint64_t vaddr = (uint64_t)(uintptr_t)(Vt + (size_t)(b * S_LEN));

  // Prologue: stage block 0 into buffer 0
  if (wave == 0) {
    tdm_load_2d(&ldsK[0][0], kaddr, DMODEL, S_LEN, DMODEL, 32, DMODEL);
    tdm_load_2d(&ldsV[0][0], vaddr, BS, DMODEL, 32, DMODEL, BS);
  }

  for (int it = 0; it < nblk; ++it) {
    const int t0  = it * 32;
    const int buf = it & 1;

    if (wave == 0) wait_tensorcnt0();
    __syncthreads();  // data[buf] ready AND buf^1 fully consumed by all waves

    if (wave == 0 && it + 1 < nblk) {
      const uint64_t off = (uint64_t)(t0 + 32) * 2;
      tdm_load_2d(&ldsK[buf ^ 1][0], kaddr + off * DMODEL,
                  DMODEL, S_LEN, DMODEL, 32, DMODEL);
      tdm_load_2d(&ldsV[buf ^ 1][0], vaddr + off,
                  BS, DMODEL, 32, DMODEL, BS);
    }

    // --- scores: two 16x16 tiles over keys [t0,t0+16), [t0+16,t0+32) ---
    f32x8 s0 = {}, s1 = {};
#pragma unroll
    for (int c = 0; c < 8; ++c)
      s0 = wmma_bf16(qf[c], ld_frag(&ldsK[buf][lo * DMODEL + c * 32 + hi * 16], 8), s0);
#pragma unroll
    for (int c = 0; c < 8; ++c)
      s1 = wmma_bf16(qf[c], ld_frag(&ldsK[buf][(16 + lo) * DMODEL + c * 32 + hi * 16], 8), s1);

    // --- decay mask gamma^(i-j) (i>=j else 0); bf16 16x32 tile into LDS ---
    const int dbase = i0 - t0;
    const float bs0 = exp2f((float)dbase * L2G);
    const float bs1 = exp2f((float)(dbase - 16) * L2G);
#pragma unroll
    for (int r = 0; r < 8; ++r) {
      const int M  = r + hi * 8;
      const int d0 = dbase + M - lo;
      const float w0 = (d0 >= 0)  ? bs0 * ld8[r] : 0.0f;
      const float w1 = (d0 >= 16) ? bs1 * ld8[r] : 0.0f;
      sc[wave][M * 32 + lo]      = (__bf16)(s0[r] * w0);
      sc[wave][M * 32 + 16 + lo] = (__bf16)(s1[r] * w1);
    }
    asm volatile("s_wait_dscnt 0x0" ::: "memory");  // per-wave LDS RAW fence

    // Scores in A-matrix layout (M = lo, K = hi*16 + {0..15})
    bf16x16 af = ld_frag(&sc[wave][lo * 32 + hi * 16], 8);

    // --- out += scores @ V : B[k][n] = V^T_lds[n*16+lo][hi*16+e] ---
#pragma unroll
    for (int n = 0; n < 16; ++n)
      acc[n] = wmma_bf16(af, ld_frag(&ldsV[buf][(n * 16 + lo) * 32 + hi * 16], 8), acc[n]);
  }

  float* op = out + (size_t)(b * S_LEN + i0) * DMODEL;
#pragma unroll
  for (int n = 0; n < 16; ++n)
#pragma unroll
    for (int r = 0; r < 8; ++r)
      op[(size_t)(r + hi * 8) * DMODEL + n * 16 + lo] = acc[n][r];
}

// ---------------------------------------------------------------------------
extern "C" void kernel_launch(void* const* d_in, const int* in_sizes, int n_in,
                              void* d_out, int out_size, void* d_ws, size_t ws_size,
                              hipStream_t stream) {
  const float* x  = (const float*)d_in[0];
  const float* Wq = (const float*)d_in[1];
  const float* Wk = (const float*)d_in[2];
  const float* Wv = (const float*)d_in[3];
  float* out = (float*)d_out;

  // Workspace: Q (8MB) | K (8MB) | V^T (8MB), all bf16 (24 MB total).
  __bf16* Qb = (__bf16*)d_ws;
  __bf16* Kb = Qb + (size_t)BS * DMODEL;
  __bf16* Vt = Kb + (size_t)BS * DMODEL;

  const int proj_waves = 3 * (BS / 16) * (DMODEL / 16);
  retnet_proj<<<proj_waves / 4, 128, 0, stream>>>(x, Wq, Wk, Wv, Qb, Kb, Vt);
  retnet_attn<<<BATCH * (S_LEN / 64), 128, 0, stream>>>(Qb, Kb, Vt, out);
}
